// skipGram_64854006170109
// MI455X (gfx1250) — compile-verified, matrix operand order checked by reading, not run on hardware
//
#include <hip/hip_runtime.h>
#include <hip/hip_bf16.h>

// ---------------------------------------------------------------------------
// Hierarchical-softmax skip-gram forward on MI455X (gfx1250, wave32).
//
//   emb    = x[1024,50000] @ W.T[50000,256] + b          (26.2 GFLOP GEMM)
//   logits = einsum('ble,be->bl', pv, emb) * (2*sign-1)
//   out    = prod_l sigmoid(logits)                       [1024]
//
// GEMM uses V_WMMA_F32_16X16X32_BF16 with a bf16 hi/lo split of both
// operands (3 WMMA passes: hh + hl + lh) for fp32-grade accuracy while
// staying at the HBM roofline (~12 us for ~256 MB of traffic).
// Split-K (16-way) + fp32 global atomics give 256 blocks / 2048 waves.
// WMMA issue order is pass-major so consecutive WMMAs hit different
// accumulators (8 independent ops between reuses -> no hazard NOPs).
// ---------------------------------------------------------------------------

typedef __attribute__((ext_vector_type(16))) __bf16 v16bf;
typedef __attribute__((ext_vector_type(8)))  float  v8f;

#define GM 1024            // batch
#define GK 50000           // vocab (K dim)
#define GN 256             // embedding (N dim)
#define GL 17              // Huffman path length
#define KSTEP 32
#define TOTAL_STEPS ((GK + KSTEP - 1) / KSTEP)    // 1563; last step has 16 valid K
#define NSPLIT 16
#define SPS ((TOTAL_STEPS + NSPLIT - 1) / NSPLIT) // 98 K-steps per split

#define WMMA_BF16(A, B, Cacc) \
  __builtin_amdgcn_wmma_f32_16x16x32_bf16(false, (A), false, (B), (short)0, (Cacc), false, false)

// Split 16 contiguous fp32 values into bf16 hi + bf16 lo fragments.
// hi = RNE(bf16(v)); residual v - float(hi) is exact in fp32; lo = bf16(residual).
// clang lowers this to v_cvt_pk_bf16_f32 + v_fma_mix_f32_bf16 pairs.
__device__ __forceinline__ void cvt_split16(const float* __restrict__ f, v16bf& hi, v16bf& lo) {
#pragma unroll
  for (int i = 0; i < 16; ++i) {
    float v = f[i];
    __bf16 h = (__bf16)v;
    hi[i] = h;
    lo[i] = (__bf16)(v - (float)h);
  }
}

__device__ __forceinline__ void load_frag(const float* __restrict__ p, v16bf& hi, v16bf& lo) {
  float f[16];
#pragma unroll
  for (int j = 0; j < 4; ++j) *(float4*)(f + 4 * j) = ((const float4*)p)[j];  // global_load_b128
  cvt_split16(f, hi, lo);
}

// Guarded variant for the single ragged K-step (K tail of 16). kAbs = absolute
// K index of f[0]; loads are 16B-aligned and GK % 16 == 0, so a base check
// per float4 is exact.
__device__ __forceinline__ void load_frag_tail(const float* __restrict__ p, int kAbs,
                                               v16bf& hi, v16bf& lo) {
  float f[16];
#pragma unroll
  for (int j = 0; j < 4; ++j) {
    float4 v;
    if (kAbs + 4 * j < GK) v = ((const float4*)p)[j];
    else { v.x = 0.f; v.y = 0.f; v.z = 0.f; v.w = 0.f; }
    *(float4*)(f + 4 * j) = v;
  }
  cvt_split16(f, hi, lo);
}

// Pass-major MMA stage: hh over all tiles, then hl, then lh. Consecutive
// WMMAs never share an accumulator, so the scheduler has 7 independent
// instructions to cover every WMMA->WMMA hazard window (no v_nop padding).
__device__ __forceinline__ void mma_stage(const v16bf ah[2], const v16bf al[2],
                                          const v16bf bh[4], const v16bf bl[4],
                                          v8f acc[2][4]) {
#pragma unroll
  for (int i = 0; i < 2; ++i)
#pragma unroll
    for (int j = 0; j < 4; ++j) acc[i][j] = WMMA_BF16(ah[i], bh[j], acc[i][j]);  // hi*hi
#pragma unroll
  for (int i = 0; i < 2; ++i)
#pragma unroll
    for (int j = 0; j < 4; ++j) acc[i][j] = WMMA_BF16(ah[i], bl[j], acc[i][j]);  // hi*lo
#pragma unroll
  for (int i = 0; i < 2; ++i)
#pragma unroll
    for (int j = 0; j < 4; ++j) acc[i][j] = WMMA_BF16(al[i], bh[j], acc[i][j]);  // lo*hi
}

// ---------------------------------------------------------------------------
// Kernel 0: seed the accumulator with the bias:  C[b,e] = bias[e]
// ---------------------------------------------------------------------------
__global__ void hs_init_kernel(const float* __restrict__ bias, float* __restrict__ C) {
  int i = blockIdx.x * blockDim.x + threadIdx.x;   // 0 .. GM*GN-1
  C[i] = bias[i & (GN - 1)];
}

// ---------------------------------------------------------------------------
// Kernel 1: split-K GEMM, C += x @ W.T (partials via fp32 global atomics).
// Block = 256 threads = 8 waves (4 in M x 2 in N); block tile 128M x 128N;
// wave tile 32M x 64N = 2x4 WMMA subtiles.
// ---------------------------------------------------------------------------
__global__ __launch_bounds__(256, 1)
void hs_gemm_kernel(const float* __restrict__ X, const float* __restrict__ W,
                    float* __restrict__ C) {
  const int lane  = threadIdx.x & 31;
  const int wave  = threadIdx.x >> 5;
  const int mw    = wave >> 1;             // 0..3
  const int nw    = wave & 1;              // 0..1
  const int r     = lane & 15;             // row/col within 16-tile
  const int khalf = (lane >> 4) << 4;      // 0 or 16: K-half owned by this lane

  const int mBase = blockIdx.x * 128 + mw * 32;
  const int nBase = blockIdx.y * 128 + nw * 64;

  // Per-lane base pointers: lane owns 16 contiguous K values of one row.
  const float* xrow[2];
#pragma unroll
  for (int i = 0; i < 2; ++i) xrow[i] = X + (size_t)(mBase + i * 16 + r) * GK + khalf;
  const float* wrow[4];
#pragma unroll
  for (int j = 0; j < 4; ++j) wrow[j] = W + (size_t)(nBase + j * 16 + r) * GK + khalf;

  v8f acc[2][4];
#pragma unroll
  for (int i = 0; i < 2; ++i)
#pragma unroll
    for (int j = 0; j < 4; ++j)
#pragma unroll
      for (int q = 0; q < 8; ++q) acc[i][j][q] = 0.0f;

  const int sBeg  = blockIdx.z * SPS;
  int sEnd        = sBeg + SPS;
  if (sEnd > TOTAL_STEPS) sEnd = TOTAL_STEPS;
  int sFull       = sEnd < (TOTAL_STEPS - 1) ? sEnd : (TOTAL_STEPS - 1);

  for (int s = sBeg; s < sFull; ++s) {
    const int k = s * KSTEP;
    v16bf ah[2], al[2], bh[4], bl[4];
#pragma unroll
    for (int i = 0; i < 2; ++i) load_frag(xrow[i] + k, ah[i], al[i]);
#pragma unroll
    for (int j = 0; j < 4; ++j) load_frag(wrow[j] + k, bh[j], bl[j]);
    mma_stage(ah, al, bh, bl, acc);
  }

  if (sEnd == TOTAL_STEPS) {               // only the last split owns the ragged step
    const int k = (TOTAL_STEPS - 1) * KSTEP;
    v16bf ah[2], al[2], bh[4], bl[4];
#pragma unroll
    for (int i = 0; i < 2; ++i) load_frag_tail(xrow[i] + k, k + khalf, ah[i], al[i]);
#pragma unroll
    for (int j = 0; j < 4; ++j) load_frag_tail(wrow[j] + k, k + khalf, bh[j], bl[j]);
    mma_stage(ah, al, bh, bl, acc);
  }

  // C/D layout: VGPR q holds M = q + (lane>>4)*8, N = lane&15 within the tile.
  const int rowHalf = (lane >> 4) * 8;
  const int colLane = lane & 15;
#pragma unroll
  for (int i = 0; i < 2; ++i) {
#pragma unroll
    for (int j = 0; j < 4; ++j) {
      const int row = mBase + i * 16 + rowHalf;
      const int col = nBase + j * 16 + colLane;
#pragma unroll
      for (int q = 0; q < 8; ++q)
        atomicAdd(&C[(size_t)(row + q) * GN + col], acc[i][j][q]);
    }
  }
}

// ---------------------------------------------------------------------------
// Kernel 2: per-batch path dots, sign, sigmoid, product. One 256-thread
// block per batch row; LDS tree reduction per path node.
// ---------------------------------------------------------------------------
__global__ __launch_bounds__(256)
void hs_logits_kernel(const float* __restrict__ C, const float* __restrict__ pv,
                      const int* __restrict__ ps, float* __restrict__ out) {
  __shared__ float red[256];
  const int b = blockIdx.x;
  const int t = threadIdx.x;
  const float e = C[(size_t)b * GN + t];

  float prod = 1.0f;
  for (int l = 0; l < GL; ++l) {
    red[t] = pv[((size_t)b * GL + l) * GN + t] * e;
    __syncthreads();
#pragma unroll
    for (int s = 128; s > 0; s >>= 1) {
      if (t < s) red[t] += red[t + s];
      __syncthreads();
    }
    if (t == 0) {
      const float sgn   = (float)(2 * ps[b * GL + l] - 1);
      const float logit = red[0] * sgn;
      prod *= 1.0f / (1.0f + expf(-logit));
    }
    __syncthreads();
  }
  if (t == 0) out[b] = prod;
}

// ---------------------------------------------------------------------------
extern "C" void kernel_launch(void* const* d_in, const int* in_sizes, int n_in,
                              void* d_out, int out_size, void* d_ws, size_t ws_size,
                              hipStream_t stream) {
  (void)in_sizes; (void)n_in; (void)out_size; (void)ws_size;
  const float* X    = (const float*)d_in[0];   // [1024, 50000]
  const float* W    = (const float*)d_in[1];   // [256, 50000]
  const float* bias = (const float*)d_in[2];   // [256]
  const float* pv   = (const float*)d_in[3];   // [1024, 17, 256]
  const int*   ps   = (const int*)d_in[4];     // [1024, 17]
  float* out        = (float*)d_out;           // [1024]
  float* C          = (float*)d_ws;            // [1024, 256] fp32 accumulator (1 MiB)

  hs_init_kernel<<<dim3(GM * GN / 256), dim3(256), 0, stream>>>(bias, C);
  hs_gemm_kernel<<<dim3(GM / 128, GN / 128, NSPLIT), dim3(256), 0, stream>>>(X, W, C);
  hs_logits_kernel<<<dim3(GM), dim3(256), 0, stream>>>(C, pv, ps, out);
}
